// LFGA_30975304139544
// MI455X (gfx1250) — matrix operand order repaired and checked by reading, not compiled, not run on hardware
//
#include <hip/hip_runtime.h>

// ---------------------------------------------------------------------------
// Problem constants (from the reference)
// ---------------------------------------------------------------------------
static constexpr int kB  = 8;
static constexpr int kC  = 256;
static constexpr int kCQ = 64;
static constexpr int kN  = 4096;        // H*W
static constexpr int kNT = 64;          // key tile / position tile
static constexpr int kMQ = 128;         // queries per attention block (8 waves x 16 rows)
static constexpr int kPAD = 264;        // padded row length (channels) for proj LDS tile

typedef __attribute__((ext_vector_type(16))) __bf16 v16bf;
typedef __attribute__((ext_vector_type(8)))  float  v8f;

union FragB {
    v16bf v;
    uint4 q[2];
    unsigned short s[16];
};

__device__ __forceinline__ unsigned short f2bf(float f) {
    union { float f; unsigned u; } x; x.f = f;
    unsigned r = x.u + 0x7FFFu + ((x.u >> 16) & 1u);   // round-to-nearest-even
    return (unsigned short)(r >> 16);
}

// ---------------------------------------------------------------------------
// Kernel 0: convert weights f32 -> bf16 (row-major [o][c])
// ---------------------------------------------------------------------------
__global__ void cvt_weights(const float* __restrict__ Wq, const float* __restrict__ Wk,
                            const float* __restrict__ Wv,
                            unsigned short* __restrict__ Wqb, unsigned short* __restrict__ Wkb,
                            unsigned short* __restrict__ Wvb) {
    int i = blockIdx.x * blockDim.x + threadIdx.x;
    if (i < kCQ * kC) { Wqb[i] = f2bf(Wq[i]); Wkb[i] = f2bf(Wk[i]); }
    if (i < kC * kC)  { Wvb[i] = f2bf(Wv[i]); }
}

// ---------------------------------------------------------------------------
// WMMA tile helper: out(16x16) = W[m*16.., :256] x X[:, nt*16..] (K = 256, 8 steps)
// A layout: lane row = l%16, K chunks {c0..c0+7, c0+16..c0+23}, c0 = s*32 + (h?8:0)
// B layout: lane col = l%16, K chunks {c0..c0+7, c0+16..c0+23}, c0 = s*32 + (h?8:0)
// ---------------------------------------------------------------------------
__device__ __forceinline__ v8f gemm_tile_256(const unsigned short* __restrict__ Wmat,
                                             const unsigned short* sX,
                                             int m, int nt, int lane) {
    const int jn = lane & 15;
    v8f acc = {0.f,0.f,0.f,0.f,0.f,0.f,0.f,0.f};
#pragma unroll
    for (int s = 0; s < 8; ++s) {
        FragB A, Bf;
        const int c0a = s * 32 + ((lane < 16) ? 0 : 8);
        const unsigned short* wrow = Wmat + (size_t)(m * 16 + jn) * kC + c0a;
        A.q[0] = *(const uint4*)(wrow);
        A.q[1] = *(const uint4*)(wrow + 16);
        const int c0b = s * 32 + ((lane < 16) ? 0 : 8);
        const unsigned short* xrow = sX + (nt * 16 + jn) * kPAD + c0b;
        Bf.q[0] = *(const uint4*)(xrow);
        Bf.q[1] = *(const uint4*)(xrow + 16);
        acc = __builtin_amdgcn_wmma_f32_16x16x32_bf16(false, A.v, false, Bf.v,
                                                      (short)0, acc, false, false);
    }
    return acc;
}

// ---------------------------------------------------------------------------
// Kernel 1: 1x1-conv projections.
//   Q,K from fb -> Qg/Kg  bf16 [b][n][64]   (row-major in d)
//   V   from fa -> Vtg    bf16 [b][c][4096] (transposed: channel-major)
// One block = one (batch, 64-position tile). 256 threads = 8 waves.
// ---------------------------------------------------------------------------
__global__ __launch_bounds__(256) void proj_kernel(
    const float* __restrict__ fa, const float* __restrict__ fb,
    const unsigned short* __restrict__ Wqb, const unsigned short* __restrict__ Wkb,
    const unsigned short* __restrict__ Wvb,
    const float* __restrict__ bq, const float* __restrict__ bk,
    const float* __restrict__ bv,
    unsigned short* __restrict__ Qg, unsigned short* __restrict__ Kg,
    unsigned short* __restrict__ Vtg)
{
    __shared__ __align__(16) unsigned short sX[kNT * kPAD];  // [n][c] bf16 tile

    const int b    = blockIdx.y;
    const int n0   = blockIdx.x * kNT;
    const int tid  = threadIdx.x;
    const int lane = tid & 31;
    const int wave = tid >> 5;
    const int jn   = lane & 15;
    const int h    = lane >> 4;

    // ---- Phase 1: stage fb tile (transposed, bf16), compute Q and K --------
    {
        const float* p = fb + ((size_t)(b * kC + tid)) * kN + n0;   // thread owns channel c=tid
#pragma unroll 4
        for (int n = 0; n < kNT; n += 4) {
            float4 v4 = *(const float4*)(p + n);
            sX[(n + 0) * kPAD + tid] = f2bf(v4.x);
            sX[(n + 1) * kPAD + tid] = f2bf(v4.y);
            sX[(n + 2) * kPAD + tid] = f2bf(v4.z);
            sX[(n + 3) * kPAD + tid] = f2bf(v4.w);
        }
    }
    __syncthreads();

    for (int i = 0; i < 4; ++i) {                 // 32 Q/K tiles over 8 waves
        const int t   = wave * 4 + i;             // 0..31
        const bool isQ = (t < 16);
        const int tt  = t & 15;
        const int m   = tt >> 2;                  // 0..3  (o tile)
        const int nt  = tt & 3;                   // 0..3  (n tile)
        const unsigned short* Wmat = isQ ? Wqb : Wkb;
        const float* bias          = isQ ? bq  : bk;
        unsigned short* dst        = isQ ? Qg  : Kg;

        v8f acc = gemm_tile_256(Wmat, sX, m, nt, lane);

        union { unsigned short s[8]; uint4 q; } pk;
#pragma unroll
        for (int j = 0; j < 8; ++j) {
            const int o = m * 16 + 8 * h + j;     // C layout: row = j + 8h
            pk.s[j] = f2bf(acc[j] + bias[o]);
        }
        const int n = n0 + nt * 16 + jn;
        *(uint4*)(dst + ((size_t)(b * kN + n)) * kCQ + m * 16 + 8 * h) = pk.q;
    }
    __syncthreads();

    // ---- Phase 2: stage fa tile, compute V (stored transposed) -------------
    {
        const float* p = fa + ((size_t)(b * kC + tid)) * kN + n0;
#pragma unroll 4
        for (int n = 0; n < kNT; n += 4) {
            float4 v4 = *(const float4*)(p + n);
            sX[(n + 0) * kPAD + tid] = f2bf(v4.x);
            sX[(n + 1) * kPAD + tid] = f2bf(v4.y);
            sX[(n + 2) * kPAD + tid] = f2bf(v4.z);
            sX[(n + 3) * kPAD + tid] = f2bf(v4.w);
        }
    }
    __syncthreads();

    for (int i = 0; i < 8; ++i) {                 // 64 V tiles over 8 waves
        const int t  = wave * 8 + i;              // 0..63
        const int m  = t >> 2;                    // 0..15
        const int nt = t & 3;                     // 0..3

        v8f acc = gemm_tile_256(Wvb, sX, m, nt, lane);

        const int n = n0 + nt * 16 + jn;
#pragma unroll
        for (int j = 0; j < 8; ++j) {
            const int o = m * 16 + 8 * h + j;
            Vtg[((size_t)(b * kC + o)) * kN + n] = f2bf(acc[j] + bv[o]);
        }
    }
}

// ---------------------------------------------------------------------------
// Async stage of one V^T tile (64 keys x 256 ch, bf16) into LDS via the
// CDNA5 async copy path (GLOBAL_LOAD_ASYNC_TO_LDS_B128, tracked by ASYNCcnt).
// Uniform 8 iterations per thread; each lane moves 16B chunks directly
// global -> LDS without touching VGPRs.
// ---------------------------------------------------------------------------
__device__ __forceinline__ void async_stage_vt(const unsigned short* __restrict__ src,
                                               unsigned short* dstLds, int tid) {
#pragma unroll
    for (int r = 0; r < 8; ++r) {
        const int idx = tid + r * 256;            // 0..2047
        const int ch = idx >> 3, ck = idx & 7;
        const unsigned lds_addr =
            (unsigned)(unsigned long long)(dstLds + ch * kNT + ck * 8);
        const unsigned long long gaddr =
            (unsigned long long)(src + (size_t)ch * kN + ck * 8);
        asm volatile("global_load_async_to_lds_b128 %0, %1, off"
                     :: "v"(lds_addr), "v"(gaddr) : "memory");
    }
}

// ---------------------------------------------------------------------------
// Kernel 2: flash-attention style  softmax(Q K^T) V  + relu(gamma*out + fa).
// 256 threads = 8 waves; each wave owns 16 query rows x all 256 channels.
// V^T tiles double-buffered in LDS via async copies.
// ---------------------------------------------------------------------------
__global__ __launch_bounds__(256) void attn_kernel(
    const unsigned short* __restrict__ Qg, const unsigned short* __restrict__ Kg,
    const unsigned short* __restrict__ Vtg, const float* __restrict__ fa,
    const float* __restrict__ gamma, float* __restrict__ out)
{
    __shared__ __align__(16) unsigned short sVt[2][kC * kNT]; // V^T tiles, 2x32KB
    __shared__ __align__(16) unsigned short sP[kMQ * kNT];    // P tiles [row][key] 16KB

    const int b    = blockIdx.y;
    const int q0   = blockIdx.x * kMQ;
    const int tid  = threadIdx.x;
    const int lane = tid & 31;
    const int wave = tid >> 5;
    const int jn   = lane & 15;
    const int h    = lane >> 4;

    // Q A-fragments (resident): lane row = q0 + wave*16 + jn, d chunks per layout
    FragB Qf[2];
    {
        const unsigned short* qrow = Qg + ((size_t)(b * kN + q0 + wave * 16 + jn)) * kCQ;
#pragma unroll
        for (int s = 0; s < 2; ++s) {
            const int c0 = s * 32 + (h ? 8 : 0);
            Qf[s].q[0] = *(const uint4*)(qrow + c0);
            Qf[s].q[1] = *(const uint4*)(qrow + c0 + 16);
        }
    }

    v8f Oacc[16];
#pragma unroll
    for (int ct = 0; ct < 16; ++ct) Oacc[ct] = (v8f){0.f,0.f,0.f,0.f,0.f,0.f,0.f,0.f};
    float mrow[8], lrow[8];
#pragma unroll
    for (int j = 0; j < 8; ++j) { mrow[j] = -3.0e38f; lrow[j] = 0.f; }

    const unsigned short* vbase = Vtg + ((size_t)b * kC) * kN;

    // prologue: kick off async copy of the first V^T tile
    async_stage_vt(vbase, &sVt[0][0], tid);

    constexpr int kTiles = kN / kNT;
    for (int kt = 0; kt < kTiles; ++kt) {
        const int k0  = kt * kNT;
        const int cur = kt & 1;

        // our async writes into sVt[cur] have landed; publish across waves
        asm volatile("s_wait_asynccnt 0x0" ::: "memory");
        __syncthreads();

        // start DMA of the next tile into the other buffer while we compute
        if (kt + 1 < kTiles)
            async_stage_vt(vbase + (k0 + kNT), &sVt[cur ^ 1][0], tid);

        // prefetch next K tile (64 rows x 128B) into cache from one wave
        if (wave == 0 && kt + 1 < kTiles) {
            const unsigned short* nk = Kg + ((size_t)(b * kN + k0 + kNT + 2 * lane)) * kCQ;
            __builtin_prefetch(nk, 0, 3);
            __builtin_prefetch(nk + kCQ, 0, 3);
        }

        // ---- S = Q K^T : 16 rows x 64 key columns, per wave ----------------
        v8f Sacc[4];
#pragma unroll
        for (int ct = 0; ct < 4; ++ct) {
            v8f acc = {0.f,0.f,0.f,0.f,0.f,0.f,0.f,0.f};
#pragma unroll
            for (int s = 0; s < 2; ++s) {
                FragB Kf;
                const unsigned short* krow =
                    Kg + ((size_t)(b * kN + k0 + ct * 16 + jn)) * kCQ + s * 32 + (h ? 8 : 0);
                Kf.q[0] = *(const uint4*)(krow);
                Kf.q[1] = *(const uint4*)(krow + 16);
                acc = __builtin_amdgcn_wmma_f32_16x16x32_bf16(false, Qf[s].v, false, Kf.v,
                                                              (short)0, acc, false, false);
            }
            Sacc[ct] = acc;
        }

        // ---- online softmax (row stats live in the 16-lane halves) ---------
#pragma unroll
        for (int j = 0; j < 8; ++j) {
            float mx = fmaxf(fmaxf(Sacc[0][j], Sacc[1][j]), fmaxf(Sacc[2][j], Sacc[3][j]));
#pragma unroll
            for (int d = 1; d < 16; d <<= 1) mx = fmaxf(mx, __shfl_xor(mx, d, 32));
            const float mnew  = fmaxf(mrow[j], mx);
            const float alpha = __expf(mrow[j] - mnew);
            float rs = 0.f;
#pragma unroll
            for (int ct = 0; ct < 4; ++ct) {
                const float p = __expf(Sacc[ct][j] - mnew);
                Sacc[ct][j] = p;
                rs += p;
            }
#pragma unroll
            for (int d = 1; d < 16; d <<= 1) rs += __shfl_xor(rs, d, 32);
            lrow[j] = lrow[j] * alpha + rs;
            mrow[j] = mnew;
#pragma unroll
            for (int ct = 0; ct < 16; ++ct) Oacc[ct][j] *= alpha;   // rescale O
        }

        // ---- P: C-layout regs -> bf16 -> wave-private LDS -> A-layout ------
        {
            unsigned short* pw = &sP[wave * 16 * kNT];
#pragma unroll
            for (int ct = 0; ct < 4; ++ct)
#pragma unroll
                for (int j = 0; j < 8; ++j)
                    pw[(j + 8 * h) * kNT + ct * 16 + jn] = f2bf(Sacc[ct][j]);
        }
        // same-wave LDS ordering: wait for the ds stores before rereading
        asm volatile("s_wait_dscnt 0" ::: "memory");

        FragB Pf[2];
#pragma unroll
        for (int s = 0; s < 2; ++s) {
            const unsigned short* prow =
                &sP[wave * 16 * kNT + jn * kNT + s * 32 + (h ? 8 : 0)];
            Pf[s].q[0] = *(const uint4*)(prow);
            Pf[s].q[1] = *(const uint4*)(prow + 16);
        }

        // ---- O += P x V : 16 channel tiles x 2 k-steps ---------------------
#pragma unroll
        for (int ct = 0; ct < 16; ++ct) {
#pragma unroll
            for (int s = 0; s < 2; ++s) {
                FragB Vf;
                const unsigned short* vrow =
                    &sVt[cur][(ct * 16 + jn) * kNT + s * 32 + (h ? 8 : 0)];
                Vf.q[0] = *(const uint4*)(vrow);
                Vf.q[1] = *(const uint4*)(vrow + 16);
                Oacc[ct] = __builtin_amdgcn_wmma_f32_16x16x32_bf16(false, Pf[s].v, false, Vf.v,
                                                                   (short)0, Oacc[ct], false, false);
            }
        }
    }

    // ---- epilogue: out = relu(gamma * (O / l) + fa) ------------------------
    const float g = gamma[0];
#pragma unroll
    for (int j = 0; j < 8; ++j) {
        const int qidx = q0 + wave * 16 + 8 * h + j;          // C layout row
        const float rl = 1.0f / lrow[j];
#pragma unroll
        for (int ct = 0; ct < 16; ++ct) {
            const int ch = ct * 16 + jn;
            const size_t off = ((size_t)(b * kC + ch)) * kN + qidx;
            const float r = g * (Oacc[ct][j] * rl) + fa[off];
            out[off] = fmaxf(r, 0.f);
        }
    }
}

// ---------------------------------------------------------------------------
// Host-side launcher
// ---------------------------------------------------------------------------
extern "C" void kernel_launch(void* const* d_in, const int* in_sizes, int n_in,
                              void* d_out, int out_size, void* d_ws, size_t ws_size,
                              hipStream_t stream) {
    (void)in_sizes; (void)n_in; (void)out_size; (void)ws_size;

    const float* fa    = (const float*)d_in[0];
    const float* fb    = (const float*)d_in[1];
    const float* Wq    = (const float*)d_in[2];
    const float* bq    = (const float*)d_in[3];
    const float* Wk    = (const float*)d_in[4];
    const float* bk    = (const float*)d_in[5];
    const float* Wv    = (const float*)d_in[6];
    const float* bv    = (const float*)d_in[7];
    const float* gamma = (const float*)d_in[8];
    float* out = (float*)d_out;

    // Workspace layout (bf16 = unsigned short), total ~25.4 MB
    unsigned short* Qg  = (unsigned short*)d_ws;               // [B][N][64]
    unsigned short* Kg  = Qg  + (size_t)kB * kN * kCQ;         // [B][N][64]
    unsigned short* Vtg = Kg  + (size_t)kB * kN * kCQ;         // [B][C][N]
    unsigned short* Wqb = Vtg + (size_t)kB * kC * kN;          // [64][256]
    unsigned short* Wkb = Wqb + (size_t)kCQ * kC;              // [64][256]
    unsigned short* Wvb = Wkb + (size_t)kCQ * kC;              // [256][256]

    cvt_weights<<<256, 256, 0, stream>>>(Wq, Wk, Wv, Wqb, Wkb, Wvb);
    proj_kernel<<<dim3(kN / kNT, kB), 256, 0, stream>>>(fa, fb, Wqb, Wkb, Wvb,
                                                        bq, bk, bv, Qg, Kg, Vtg);
    attn_kernel<<<dim3(kN / kMQ, kB), 256, 0, stream>>>(Qg, Kg, Vtg, fa, gamma, out);
}
